// VBrNerf_36696200577474
// MI455X (gfx1250) — matrix operand order repaired and checked by reading, not compiled, not run on hardware
//
#include <hip/hip_runtime.h>
#include <hip/hip_bf16.h>

typedef __attribute__((ext_vector_type(16))) _Float16 v16h;
typedef __attribute__((ext_vector_type(8)))  _Float16 v8h;
typedef __attribute__((ext_vector_type(8)))  float    v8f;
typedef __attribute__((ext_vector_type(4)))  float    v4f;

#define RESO     128
#define NUM_FEAT 16
#define RADIUS   1.3f
#define MLP_W    128
#define BATCHN   1024
#define STEPF    (RADIUS / 128.0f)
#define NS       442            /* live samples per ray */
#define SPAD     448            /* padded to 28*16 */
#define KPAD     64             /* 40 features padded to 2 K-tiles of 32 */
#define NTILES   28

static constexpr size_t XBUF_BYTES = (size_t)BATCHN * SPAD * KPAD * sizeof(_Float16); // ~56MB

__device__ __forceinline__ unsigned expand_bits(unsigned v) {
    v = (v * 65537u) & 0xFF0000FFu;
    v = (v * 257u)   & 0x0F00F00Fu;
    v = (v * 17u)    & 0xC30C30C3u;
    v = (v * 5u)     & 0x49249249u;
    return v;
}

// ---------------------------------------------------------------------------
// Kernel 1: per-sample feature build (ray march + VQ gather + trilinear + PE)
// Writes x[64] f16 (K-padded) per sample + mask.
// ---------------------------------------------------------------------------
__global__ __launch_bounds__(256) void vbr_feat_kernel(
    const float* __restrict__ rays_o, const float* __restrict__ rays_d,
    const float* __restrict__ grid,   const float* __restrict__ codebook,
    _Float16* __restrict__ xbuf, float* __restrict__ maskbuf)
{
    int gid = blockIdx.x * 256 + threadIdx.x;
    if (gid >= BATCHN * SPAD) return;
    int ray = gid / SPAD;
    int j   = gid - ray * SPAD;

    _Float16 row[KPAD];
#pragma unroll
    for (int i = 0; i < KPAD; ++i) row[i] = (_Float16)0.f;

    float ox = rays_o[ray*3+0], oy = rays_o[ray*3+1], oz = rays_o[ray*3+2];
    float dx = rays_d[ray*3+0], dy = rays_d[ray*3+1], dz = rays_d[ray*3+2];

    bool mask = false;
    if (j < NS) {
        float sdx = (fabsf(dx) < 1e-8f) ? 1e-8f : dx;
        float sdy = (fabsf(dy) < 1e-8f) ? 1e-8f : dy;
        float sdz = (fabsf(dz) < 1e-8f) ? 1e-8f : dz;
        float tax = (-RADIUS - ox) / sdx, tbx = (RADIUS - ox) / sdx;
        float tay = (-RADIUS - oy) / sdy, tby = (RADIUS - oy) / sdy;
        float taz = (-RADIUS - oz) / sdz, tbz = (RADIUS - oz) / sdz;
        float tmin = fmaxf(fmaxf(fminf(tax,tbx), fminf(tay,tby)), fminf(taz,tbz));
        float tmax = fminf(fminf(fmaxf(tax,tbx), fmaxf(tay,tby)), fmaxf(taz,tbz));
        float tnear = fmaxf(tmin, 0.f);
        float tmid = tnear + ((float)j + 0.5f) * STEPF;
        float px = ox + dx*tmid, py = oy + dy*tmid, pz = oz + dz*tmid;
        mask = (tmid < tmax) && (fabsf(px) <= RADIUS) && (fabsf(py) <= RADIUS)
                             && (fabsf(pz) <= RADIUS);

        float gx = (px / RADIUS + 1.f) * 64.f;
        float gy = (py / RADIUS + 1.f) * 64.f;
        float gz = (pz / RADIUS + 1.f) * 64.f;
        float flx = floorf(gx), fly = floorf(gy), flz = floorf(gz);
        float offx = gx - flx, offy = gy - fly, offz = gz - flz;
        int ix = (int)flx, iy = (int)fly, iz = (int)flz;
        float wx[2] = {1.f - offx, offx};
        float wy[2] = {1.f - offy, offy};
        float wz[2] = {1.f - offz, offz};

        float acc[16];
#pragma unroll
        for (int k = 0; k < 16; ++k) acc[k] = 0.f;

#pragma unroll
        for (int c = 0; c < 8; ++c) {
            int bx = (c >> 2) & 1, by = (c >> 1) & 1, bz = c & 1;
            int cx = ix + bx; cx = cx < 0 ? 0 : (cx > RESO-1 ? RESO-1 : cx);
            int cy = iy + by; cy = cy < 0 ? 0 : (cy > RESO-1 ? RESO-1 : cy);
            int cz = iz + bz; cz = cz < 0 ? 0 : (cz > RESO-1 ? RESO-1 : cz);
            unsigned mi = expand_bits((unsigned)cx)
                        | (expand_bits((unsigned)cy) << 1)
                        | (expand_bits((unsigned)cz) << 2);
            const v4f* gp = (const v4f*)(grid + (size_t)mi * 16);
            v4f g0 = gp[0], g1 = gp[1], g2 = gp[2], g3 = gp[3];
            float gv[16];
#pragma unroll
            for (int k = 0; k < 4; ++k) {
                gv[k] = g0[k]; gv[4+k] = g1[k]; gv[8+k] = g2[k]; gv[12+k] = g3[k];
            }
            // straight-through VQ forward == one_hot(argmax of logits)
            int best = 0; float bvv = gv[0];
#pragma unroll
            for (int k = 1; k < 16; ++k) if (gv[k] > bvv) { bvv = gv[k]; best = k; }
            float w = wx[bx] * wy[by] * wz[bz];
            const float* cb = codebook + (size_t)best * 16;
#pragma unroll
            for (int k = 0; k < 16; ++k) acc[k] += w * cb[k];
        }
#pragma unroll
        for (int k = 0; k < 16; ++k) row[k] = (_Float16)acc[k];
        // positional encoding of direction: [sinx,siny,sinz,cosx,cosy,cosz] per freq
#pragma unroll
        for (int f = 0; f < 4; ++f) {
            float sc = (float)(1 << f);
            row[16 + f*6 + 0] = (_Float16)sinf(dx * sc);
            row[16 + f*6 + 1] = (_Float16)sinf(dy * sc);
            row[16 + f*6 + 2] = (_Float16)sinf(dz * sc);
            row[16 + f*6 + 3] = (_Float16)cosf(dx * sc);
            row[16 + f*6 + 4] = (_Float16)cosf(dy * sc);
            row[16 + f*6 + 5] = (_Float16)cosf(dz * sc);
        }
    }
    v8h* dst = (v8h*)(xbuf + (size_t)gid * KPAD);
#pragma unroll
    for (int i = 0; i < 8; ++i) dst[i] = *(v8h*)&row[i*8];
    maskbuf[gid] = mask ? 1.f : 0.f;
}

// ---------------------------------------------------------------------------
// Kernel 2: per-ray WMMA MLP + alpha compositing. 1 workgroup (4 waves) / ray.
// ---------------------------------------------------------------------------
__global__ __launch_bounds__(128) void vbr_mlp_kernel(
    const float* __restrict__ rays_d,
    const float* __restrict__ W1, const float* __restrict__ b1,
    const float* __restrict__ W2, const float* __restrict__ b2,
    const _Float16* __restrict__ xbuf, const float* __restrict__ maskbuf,
    float* __restrict__ out_rgb, float* __restrict__ out_alpha)
{
    // B-fragment-swizzled weights: [ntile][ktile][lane][16 halves]
    __shared__ __align__(32) _Float16 Wf1[8 * 2 * 32 * 16];   // 16 KB
    __shared__ __align__(32) _Float16 Wf2[4 * 32 * 16];       //  4 KB
    __shared__ float b1s[MLP_W];
    __shared__ float b2s[4];
    __shared__ __align__(16) _Float16 hbuf[4][16][MLP_W];     // 16 KB, per wave
    __shared__ float preA[SPAD][4];                           //  7 KB
    __shared__ float alphaL[SPAD];
    __shared__ float colL[SPAD][3];
    __shared__ float part[32], Tpref[32], red[32][4];

    const int ray  = blockIdx.x;
    const int tid  = threadIdx.x;
    const int w    = tid >> 5;
    const int lane = tid & 31;
    const int m15  = lane & 15;
    const int hi   = (lane & 16) ? 1 : 0;
    const int kb   = hi * 8;

    // ---- swizzle W1 into B-fragment layout (k = ktile*32 + (i&8)*2 + hi*8 + (i&7)) ----
    for (int e = tid; e < 8*2*32*16; e += 128) {
        int i  = e & 15;
        int ln = (e >> 4) & 31;
        int kt = (e >> 9) & 1;
        int nt = e >> 10;
        int n  = (nt << 4) + (ln & 15);
        int k  = (kt << 5) + ((ln & 16) ? 8 : 0) + (i & 7) + ((i & 8) ? 16 : 0);
        Wf1[e] = (k < 40) ? (_Float16)W1[k * MLP_W + n] : (_Float16)0.f;
    }
    for (int e = tid; e < 4*32*16; e += 128) {
        int i  = e & 15;
        int ln = (e >> 4) & 31;
        int kt = e >> 9;
        int n  = ln & 15;
        int k  = (kt << 5) + ((ln & 16) ? 8 : 0) + (i & 7) + ((i & 8) ? 16 : 0);
        Wf2[e] = (n < 4) ? (_Float16)W2[k * 4 + n] : (_Float16)0.f;
    }
    if (tid < MLP_W) b1s[tid] = b1[tid];
    if (tid < 4)     b2s[tid] = b2[tid];
    __syncthreads();

    // ---- hoist loop-invariant layer-2 B fragments into registers ----
    v16h wf2r[4];
#pragma unroll
    for (int kt = 0; kt < 4; ++kt)
        wf2r[kt] = *(const v16h*)&Wf2[(kt * 32 + lane) * 16];

    // ---- hoist layer-1 bias values (loop-invariant) ----
    float bvs[8];
#pragma unroll
    for (int nt = 0; nt < 8; ++nt) bvs[nt] = b1s[(nt << 4) + m15];
    float bv2 = (m15 < 4) ? b2s[m15] : 0.f;

    const _Float16* xbase = xbuf + ((size_t)ray * SPAD + m15) * KPAD;

    // ---- software-pipelined WMMA MLP over 16-sample tiles ----
    // A fragments (16x32 f16): two aligned 16B global chunks per K-tile per lane
    v16h a0, a1;
    {
        const _Float16* xr = xbase + (size_t)(w * 16) * KPAD;
        v8h a0l = *(const v8h*)(xr + 0  + kb);
        v8h a0h = *(const v8h*)(xr + 16 + kb);
        v8h a1l = *(const v8h*)(xr + 32 + kb);
        v8h a1h = *(const v8h*)(xr + 48 + kb);
        a0 = __builtin_shufflevector(a0l, a0h, 0,1,2,3,4,5,6,7,8,9,10,11,12,13,14,15);
        a1 = __builtin_shufflevector(a1l, a1h, 0,1,2,3,4,5,6,7,8,9,10,11,12,13,14,15);
    }

    for (int t = w; t < NTILES; t += 4) {
        int s0 = t * 16;

        // prefetch next tile's A fragments (overlap HBM/L2 latency with WMMAs)
        v8h n0l, n0h, n1l, n1h;
        const bool have_next = (t + 4) < NTILES;
        if (have_next) {
            const _Float16* xr = xbase + (size_t)((t + 4) * 16) * KPAD;
            n0l = *(const v8h*)(xr + 0  + kb);
            n0h = *(const v8h*)(xr + 16 + kb);
            n1l = *(const v8h*)(xr + 32 + kb);
            n1h = *(const v8h*)(xr + 48 + kb);
        }

        // layer 1: x[16x64] @ W1[64x128] + b1, relu -> hbuf (f16)
#pragma unroll
        for (int nt = 0; nt < 8; ++nt) {
            float bv = bvs[nt];
            v8f c = {bv, bv, bv, bv, bv, bv, bv, bv};
            v16h bf0 = *(const v16h*)&Wf1[((nt*2 + 0) * 32 + lane) * 16];
            v16h bf1 = *(const v16h*)&Wf1[((nt*2 + 1) * 32 + lane) * 16];
            c = __builtin_amdgcn_wmma_f32_16x16x32_f16(false, a0, false, bf0,
                                                       (short)0, c, false, false);
            c = __builtin_amdgcn_wmma_f32_16x16x32_f16(false, a1, false, bf1,
                                                       (short)0, c, false, false);
#pragma unroll
            for (int r = 0; r < 8; ++r) {
                float h = fmaxf(c[r], 0.f);
                hbuf[w][r + hi*8][(nt << 4) + m15] = (_Float16)h;
            }
        }
        asm volatile("s_wait_dscnt 0" ::: "memory");

        // layer 2: h[16x128] @ W2[128x16(pad)] + b2; B frags already in regs,
        // issue all 8 hbuf loads up front so only one dscnt wait gates the chain
        v8h h2[8];
#pragma unroll
        for (int kt = 0; kt < 4; ++kt) {
            h2[kt*2+0] = *(const v8h*)&hbuf[w][m15][kt*32 + kb];
            h2[kt*2+1] = *(const v8h*)&hbuf[w][m15][kt*32 + 16 + kb];
        }
        v8f c2 = {bv2, bv2, bv2, bv2, bv2, bv2, bv2, bv2};
#pragma unroll
        for (int kt = 0; kt < 4; ++kt) {
            v16h a2 = __builtin_shufflevector(h2[kt*2+0], h2[kt*2+1],
                                              0,1,2,3,4,5,6,7,8,9,10,11,12,13,14,15);
            c2 = __builtin_amdgcn_wmma_f32_16x16x32_f16(false, a2, false, wf2r[kt],
                                                        (short)0, c2, false, false);
        }
        if (m15 < 4) {
#pragma unroll
            for (int r = 0; r < 8; ++r) preA[s0 + r + hi*8][m15] = c2[r];
        }

        // rotate pipeline
        if (have_next) {
            a0 = __builtin_shufflevector(n0l, n0h, 0,1,2,3,4,5,6,7,8,9,10,11,12,13,14,15);
            a1 = __builtin_shufflevector(n1l, n1h, 0,1,2,3,4,5,6,7,8,9,10,11,12,13,14,15);
        }
    }
    __syncthreads();

    // ---- compositing: alpha / sigmoid colors in parallel ----
    float dxr = rays_d[ray*3+0], dyr = rays_d[ray*3+1], dzr = rays_d[ray*3+2];
    float delta = STEPF * sqrtf(dxr*dxr + dyr*dyr + dzr*dzr);
    for (int s = tid; s < SPAD; s += 128) {
        float a = 0.f, c0 = 0.f, c1 = 0.f, c2v = 0.f;
        if (s < NS) {
            float msk  = maskbuf[(size_t)ray * SPAD + s];
            float dens = (msk > 0.5f) ? fmaxf(preA[s][0], 0.f) : 0.f;
            a  = 1.f - expf(-dens * delta);
            c0 = 1.f / (1.f + expf(-preA[s][1]));
            c1 = 1.f / (1.f + expf(-preA[s][2]));
            c2v= 1.f / (1.f + expf(-preA[s][3]));
            out_alpha[(size_t)ray * NS + s] = a;
        }
        alphaL[s] = a; colL[s][0] = c0; colL[s][1] = c1; colL[s][2] = c2v;
    }
    __syncthreads();

    // ---- chunked exclusive-cumprod scan + weighted accumulation (wave 0) ----
    if (tid < 32) {
        int sa = lane * 14;                  // 32*14 = 448 covers SPAD
        float P = 1.f;
#pragma unroll
        for (int q = 0; q < 14; ++q) P *= (1.f - alphaL[sa + q] + 1e-10f);
        part[lane] = P;
        asm volatile("s_wait_dscnt 0" ::: "memory");
        if (lane == 0) {
            float T = 1.f;
            for (int i = 0; i < 32; ++i) { Tpref[i] = T; T *= part[i]; }
        }
        asm volatile("s_wait_dscnt 0" ::: "memory");
        float T = Tpref[lane];
        float r0 = 0.f, r1 = 0.f, r2 = 0.f, ws = 0.f;
#pragma unroll
        for (int q = 0; q < 14; ++q) {
            int s = sa + q;
            float al = alphaL[s];
            float wt = al * T;
            r0 += wt * colL[s][0]; r1 += wt * colL[s][1]; r2 += wt * colL[s][2];
            ws += wt;
            T *= (1.f - al + 1e-10f);
        }
        red[lane][0] = r0; red[lane][1] = r1; red[lane][2] = r2; red[lane][3] = ws;
        asm volatile("s_wait_dscnt 0" ::: "memory");
        if (lane == 0) {
            float t0 = 0.f, t1 = 0.f, t2 = 0.f, tw = 0.f;
            for (int i = 0; i < 32; ++i) {
                t0 += red[i][0]; t1 += red[i][1]; t2 += red[i][2]; tw += red[i][3];
            }
            float bg = 1.f - tw;  // white background
            out_rgb[ray*3+0] = t0 + bg;
            out_rgb[ray*3+1] = t1 + bg;
            out_rgb[ray*3+2] = t2 + bg;
        }
    }
}

extern "C" void kernel_launch(void* const* d_in, const int* in_sizes, int n_in,
                              void* d_out, int out_size, void* d_ws, size_t ws_size,
                              hipStream_t stream) {
    const float* rays_o   = (const float*)d_in[0];
    const float* rays_d   = (const float*)d_in[1];
    const float* grid     = (const float*)d_in[2];
    const float* codebook = (const float*)d_in[3];
    const float* W1       = (const float*)d_in[4];
    const float* b1       = (const float*)d_in[5];
    const float* W2       = (const float*)d_in[6];
    const float* b2       = (const float*)d_in[7];
    // d_in[8] = level (always 0)

    float* out       = (float*)d_out;
    float* out_rgb   = out;                    // [1024, 3]
    float* out_alpha = out + (size_t)BATCHN*3; // [1024, 442]

    _Float16* xbuf   = (_Float16*)d_ws;
    float*    maskbuf= (float*)((char*)d_ws + XBUF_BYTES);

    int total = BATCHN * SPAD;                 // 458752 = 1792 * 256
    vbr_feat_kernel<<<total / 256, 256, 0, stream>>>(
        rays_o, rays_d, grid, codebook, xbuf, maskbuf);
    vbr_mlp_kernel<<<BATCHN, 128, 0, stream>>>(
        rays_d, W1, b1, W2, b2, xbuf, maskbuf, out_rgb, out_alpha);
}
